// PixelSample_41549513622110
// MI455X (gfx1250) — matrix operand order repaired and checked by reading, not compiled, not run on hardware
//
#include <hip/hip_runtime.h>
#include <hip/hip_bf16.h>

typedef __attribute__((ext_vector_type(16))) _Float16 v16h;
typedef __attribute__((ext_vector_type(8)))  _Float16 v8h;
typedef __attribute__((ext_vector_type(8)))  float    v8f;

#define NEG_SLOPE 0.01f

// ---------------------------------------------------------------- utilities
__global__ void k_zero_f32(float* p, int n) {
  int i = blockIdx.x * blockDim.x + threadIdx.x;
  if (i < n) p[i] = 0.f;
}

// x: [4][116*116][32] HWC-f16; ch 0..3 = global feats, 4..15 = features, 16..31 = 0
__global__ void k_pack_input(const float* __restrict__ feats,
                             const float* __restrict__ gf,
                             _Float16* __restrict__ x) {
  const int HW = 116 * 116;
  int i = blockIdx.x * blockDim.x + threadIdx.x;
  if (i >= 4 * HW * 32) return;
  int c = i & 31;
  int rest = i >> 5;
  int pix = rest % HW;
  int s = rest / HW;
  float v = 0.f;
  if (c < 4) v = gf[c];
  else if (c < 16) v = feats[(s * 12 + (c - 4)) * HW + pix];
  x[i] = (_Float16)v;
}

// OIHW f32 -> [Cout][KK][CinPad] f16 (zero-padded channels)
__global__ void k_conv_weights(const float* __restrict__ w, _Float16* __restrict__ o,
                               int Cout, int Cin, int CinPad, int KK) {
  int i = blockIdx.x * blockDim.x + threadIdx.x;
  int total = Cout * KK * CinPad;
  if (i >= total) return;
  int ci = i % CinPad;
  int t = (i / CinPad) % KK;
  int co = i / (CinPad * KK);
  float v = (ci < Cin) ? w[(co * Cin + ci) * KK + t] : 0.f;
  o[i] = (_Float16)v;
}

__global__ void k_max_spp(const _Float16* __restrict__ in, _Float16* __restrict__ out,
                          int n, int per) {
  int i = blockIdx.x * blockDim.x + threadIdx.x;
  if (i >= n) return;
  float m = (float)in[i];
#pragma unroll
  for (int s = 1; s < 4; ++s) m = fmaxf(m, (float)in[(size_t)s * per + i]);
  out[i] = (_Float16)m;
}

// ------------------------------------------------- implicit-GEMM 3x3 conv (WMMA)
// Each wave: 16 out-ch x 64 pixels (4 N-tiles, one shared A fragment per k-step).
// in : [B][Hin*Win][Cin]  f16 (HWC), Cin multiple of 32
// wt : [Cout][9][Cin]     f16
// out: [B][Hout*Wout][Cout] f16
__global__ void __launch_bounds__(256)
k_conv3x3_wmma(const _Float16* __restrict__ in,
               const _Float16* __restrict__ wt,
               const float* __restrict__ bias,
               _Float16* __restrict__ out,
               int Hin, int Win, int Cin, int Cout, int doRelu) {
  const int Hout = Hin - 2, Wout = Win - 2;
  const int HW = Hout * Wout;
  const int lane = threadIdx.x & 31;
  const int wave = threadIdx.x >> 5;
  const int obase = wave * 16;           // out-channel tile per wave
  if (obase >= Cout) return;             // wave-uniform

  const bool hi = lane >= 16;
  const int l15 = lane & 15;
  const int aOff = hi ? 8 : 0;           // D-row offset: hi lanes hold rows 8..15
  const int bOff = hi ? 16 : 0;          // A/B-frag K-split: contiguous k = 16h + s in both

  const _Float16* inS = in + (size_t)blockIdx.z * Hin * Win * Cin;
  _Float16* outS = out + (size_t)blockIdx.z * HW * Cout;

  // 4 pixel tiles of 16; this lane's column pixel for each tile
  int pp[4];
  bool valid[4];
  size_t brow[4];                        // (oy*Win + ox) * Cin
#pragma unroll
  for (int nt = 0; nt < 4; ++nt) {
    int p = blockIdx.x * 64 + nt * 16 + l15;
    valid[nt] = p < HW;
    if (p >= HW) p = HW - 1;
    pp[nt] = p;
    int oy = p / Wout, ox = p % Wout;
    brow[nt] = ((size_t)oy * Win + ox) * Cin;
  }

  v8f acc[4];
#pragma unroll
  for (int nt = 0; nt < 4; ++nt)
#pragma unroll
    for (int r = 0; r < 8; ++r) acc[nt][r] = bias[obase + r + aOff];

  for (int kc = 0; kc < Cin; kc += 32) {
#pragma unroll
    for (int tap = 0; tap < 9; ++tap) {
      const int ty = tap / 3, tx = tap % 3;
      // A: weights, row M=l15 -> out channel; two contiguous 8-half runs
      const _Float16* wp = wt + ((size_t)(obase + l15) * 9 + tap) * Cin + kc + bOff;
      v8h a0 = *(const v8h*)wp;
      v8h a1 = *(const v8h*)(wp + 8);
      v16h av;
#pragma unroll
      for (int e = 0; e < 8; ++e) { av[e] = a0[e]; av[e + 8] = a1[e]; }
      const size_t tapoff = ((size_t)ty * Win + tx) * Cin + kc + bOff;
#pragma unroll
      for (int nt = 0; nt < 4; ++nt) {
        // B: inputs, col N=l15 -> pixel; one contiguous 16-half run
        v16h bv = *(const v16h*)(inS + brow[nt] + tapoff);
        acc[nt] = __builtin_amdgcn_wmma_f32_16x16x32_f16(false, av, false, bv,
                                                         (short)0, acc[nt], false, false);
      }
    }
  }

#pragma unroll
  for (int nt = 0; nt < 4; ++nt) {
    if (doRelu) {
#pragma unroll
      for (int r = 0; r < 8; ++r)
        acc[nt][r] = acc[nt][r] > 0.f ? acc[nt][r] : acc[nt][r] * NEG_SLOPE;
    }
    if (valid[nt]) {
      v8h hv;
#pragma unroll
      for (int r = 0; r < 8; ++r) hv[r] = (_Float16)acc[nt][r];
      *(v8h*)(outS + (size_t)pp[nt] * Cout + obase + aOff) = hv;
    }
  }
}

// ------------------------------------------------- KP MLP (WMMA, rows = sites)
// lin/lout: [16][64] f16 in LDS; w: [64 out][64 in] f16; one 16x64 layer per wave
__device__ inline void kp_layer_wmma(const _Float16* lin, _Float16* lout,
                                     const _Float16* __restrict__ w,
                                     const float* __restrict__ b,
                                     int lane, bool relu) {
  const int l15 = lane & 15;
  const bool hi = lane >= 16;
  const int aOff = hi ? 8 : 0;
  const int bOff = hi ? 16 : 0;
  v16h a[2];
  const _Float16* rowp = lin + l15 * 64;   // A row M = site
#pragma unroll
  for (int c = 0; c < 2; ++c) {
    v8h a0 = *(const v8h*)(rowp + c * 32 + bOff);
    v8h a1 = *(const v8h*)(rowp + c * 32 + bOff + 8);
#pragma unroll
    for (int e = 0; e < 8; ++e) { a[c][e] = a0[e]; a[c][e + 8] = a1[e]; }
  }
#pragma unroll
  for (int n = 0; n < 4; ++n) {            // 4 out-tiles of 16
    float bb = b[n * 16 + l15];            // bias by output column
    v8f acc;
#pragma unroll
    for (int r = 0; r < 8; ++r) acc[r] = bb;
#pragma unroll
    for (int c = 0; c < 2; ++c) {
      v16h bv = *(const v16h*)(w + (size_t)(n * 16 + l15) * 64 + c * 32 + bOff);
      acc = __builtin_amdgcn_wmma_f32_16x16x32_f16(false, a[c], false, bv,
                                                   (short)0, acc, false, false);
    }
#pragma unroll
    for (int r = 0; r < 8; ++r) {
      float v = acc[r];
      if (relu) v = v > 0.f ? v : v * NEG_SLOPE;
      lout[(r + aOff) * 64 + n * 16 + l15] = (_Float16)v;
    }
  }
}

__global__ void __launch_bounds__(256)
k_kp_mlp(const float* __restrict__ feats,    // [4][12][116][116]
         const float* __restrict__ rad,      // [4][3][116][116]
         const _Float16* __restrict__ pfeat, // [96*96][64]
         const float* __restrict__ sfw,      // [64][12]
         const float* __restrict__ sfb,      // [64]
         const _Float16* __restrict__ kw0,   // [64][64] f16
         const float* __restrict__ kb0,
         const _Float16* __restrict__ kw1,
         const float* __restrict__ kb1,
         const float* __restrict__ kw2,      // [64]
         const float* __restrict__ kb2,      // [1]
         float* __restrict__ unnorm,         // [3][9216]
         float* __restrict__ sumw) {         // [9216]
  __shared__ __align__(32) _Float16 ldsV[8][16][64];
  __shared__ __align__(32) _Float16 ldsH[8][16][64];
  const int lane = threadIdx.x & 31;
  const int wave = threadIdx.x >> 5;
  const int base = (blockIdx.x * 8 + wave) * 16;   // 16 sites per wave

  const int r = lane >> 1;      // site row handled by this lane pair
  const int hs = lane & 1;      // half of the 64-vector
  const int site = base + r;    // ((y*96+x)*4 + s)*25 + dy*5 + dx
  const int td = site % 25;
  const int dy = td / 5, dx = td % 5;
  const int s = (site / 25) & 3;
  const int pix = site / 100;
  const int y = pix / 96, x = pix % 96;
  const int FHW = 116 * 116;
  const int fpos = (8 + y + dy) * 116 + (8 + x + dx);   // crop = 8

  // stage 1: row = pfeat[pixel] + sf_b + sf_w @ tap12 (tap offsets on ch 0/1)
  {
    float t[12];
#pragma unroll
    for (int f = 0; f < 12; ++f) t[f] = feats[(s * 12 + f) * FHW + fpos];
    t[0] += (float)(dx - 2);
    t[1] += (float)(dy - 2);
    const _Float16* pf = pfeat + (size_t)pix * 64 + hs * 32;
    for (int c = 0; c < 32; ++c) {
      int cc = hs * 32 + c;
      float v = (float)pf[c] + sfb[cc];
#pragma unroll
      for (int f = 0; f < 12; ++f) v += sfw[cc * 12 + f] * t[f];
      ldsV[wave][r][cc] = (_Float16)v;
    }
  }
  __syncthreads();
  kp_layer_wmma(&ldsV[wave][0][0], &ldsH[wave][0][0], kw0, kb0, lane, true);
  __syncthreads();
  kp_layer_wmma(&ldsH[wave][0][0], &ldsV[wave][0][0], kw1, kb1, lane, true);
  __syncthreads();

  // stage 4: 64 -> 1, then scatter weighted radiance
  float part = 0.f;
  {
    const _Float16* rowp = &ldsV[wave][r][hs * 32];
    for (int c = 0; c < 32; ++c) part += (float)rowp[c] * kw2[hs * 32 + c];
  }
  part += __shfl_xor(part, 1, 32);
  const float wtv = part + kb2[0];

  if (hs == 0) {
#pragma unroll
    for (int c = 0; c < 3; ++c) {
      float rv = rad[(s * 3 + c) * FHW + fpos];
      atomicAdd(&unnorm[c * 9216 + pix], wtv * rv);
    }
    atomicAdd(&sumw[pix], wtv);
  }
}

__global__ void k_finalize(const float* __restrict__ unnorm,
                           const float* __restrict__ sumw,
                           float* __restrict__ out) {
  int i = blockIdx.x * blockDim.x + threadIdx.x;
  if (i >= 3 * 9216) return;
  out[i] = unnorm[i] / (sumw[i % 9216] + 1e-8f);
}

// ---------------------------------------------------------------- launcher
extern "C" void kernel_launch(void* const* d_in, const int* in_sizes, int n_in,
                              void* d_out, int out_size, void* d_ws, size_t ws_size,
                              hipStream_t stream) {
  (void)in_sizes; (void)n_in; (void)out_size; (void)ws_size;
  const float* features = (const float*)d_in[0];
  const float* radiance = (const float*)d_in[1];
  const float* gfeat    = (const float*)d_in[2];
  const float* emb_w0   = (const float*)d_in[3];
  const float* emb_b0   = (const float*)d_in[4];
  const float* emb_w    = (const float*)d_in[5];
  const float* emb_b    = (const float*)d_in[6];
  const float* pf_w0    = (const float*)d_in[7];
  const float* pf_b0    = (const float*)d_in[8];
  const float* pf_w1    = (const float*)d_in[9];
  const float* pf_b1    = (const float*)d_in[10];
  const float* pf_w2    = (const float*)d_in[11];
  const float* pf_b2    = (const float*)d_in[12];
  const float* sf_w     = (const float*)d_in[13];
  const float* sf_b     = (const float*)d_in[14];
  const float* kp_w0    = (const float*)d_in[15];
  const float* kp_b0    = (const float*)d_in[16];
  const float* kp_w1    = (const float*)d_in[17];
  const float* kp_b1    = (const float*)d_in[18];
  const float* kp_w2    = (const float*)d_in[19];
  const float* kp_b2    = (const float*)d_in[20];
  float* out = (float*)d_out;

  char* ws = (char*)d_ws;
  auto au = [](size_t v) { return (v + 255) & ~(size_t)255; };
  size_t o = 0;
  _Float16* wX  = (_Float16*)(ws + o); o += au((size_t)4 * 13456 * 32 * 2);   // packed input / E / pf1 out
  _Float16* wA  = (_Float16*)(ws + o); o += au((size_t)4 * 12996 * 128 * 2);  // ping
  _Float16* wB  = (_Float16*)(ws + o); o += au((size_t)4 * 12544 * 128 * 2);  // pong
  _Float16* wPf = (_Float16*)(ws + o); o += au((size_t)9216 * 64 * 2);        // pfeat HWC
  _Float16* wWt = (_Float16*)(ws + o); o += au((size_t)1298432 * 2);          // all f16 weights
  float*    wUn = (float*)(ws + o);    o += au((size_t)(27648 + 9216) * 4);
  float*    wSw = wUn + 27648;

  _Float16* w0p = wWt;                 // 128*9*32
  _Float16* wE  = w0p + 36864;         // 6 * 128*9*128
  _Float16* wP0 = wE + 884736;
  _Float16* wP1 = wP0 + 147456;
  _Float16* wP2 = wP1 + 147456;        // 64*9*128
  _Float16* wK0 = wP2 + 73728;         // 64*64
  _Float16* wK1 = wK0 + 4096;

  k_zero_f32<<<(36864 + 255) / 256, 256, 0, stream>>>(wUn, 36864);
  {
    int n = 4 * 13456 * 32;
    k_pack_input<<<(n + 255) / 256, 256, 0, stream>>>(features, gfeat, wX);
  }
  auto cw = [&](const float* src, _Float16* dst, int Cout, int Cin, int CinPad, int KK) {
    int n = Cout * KK * CinPad;
    k_conv_weights<<<(n + 255) / 256, 256, 0, stream>>>(src, dst, Cout, Cin, CinPad, KK);
  };
  cw(emb_w0, w0p, 128, 16, 32, 9);
  cw(emb_w,  wE,  768, 128, 128, 9);   // all 6 layers in one pass
  cw(pf_w0,  wP0, 128, 128, 128, 9);
  cw(pf_w1,  wP1, 128, 128, 128, 9);
  cw(pf_w2,  wP2, 64, 128, 128, 9);
  cw(kp_w0,  wK0, 64, 64, 64, 1);
  cw(kp_w1,  wK1, 64, 64, 64, 1);

  auto conv = [&](const _Float16* in, const _Float16* w, const float* b,
                  _Float16* op, int Hin, int Cin, int Cout, int relu, int batch) {
    int HW = (Hin - 2) * (Hin - 2);
    dim3 grid((HW + 63) / 64, 1, batch);   // 64 pixels per wave
    k_conv3x3_wmma<<<grid, 256, 0, stream>>>(in, w, b, op, Hin, Hin, Cin, Cout, relu);
  };
  // embedding chain (batch = 4 spp samples)
  conv(wX, w0p,             emb_b0,          wA, 116, 32, 128, 1, 4);
  conv(wA, wE + 0 * 147456, emb_b + 0 * 128, wB, 114, 128, 128, 1, 4);
  conv(wB, wE + 1 * 147456, emb_b + 1 * 128, wA, 112, 128, 128, 1, 4);
  conv(wA, wE + 2 * 147456, emb_b + 2 * 128, wB, 110, 128, 128, 1, 4);
  conv(wB, wE + 3 * 147456, emb_b + 3 * 128, wA, 108, 128, 128, 1, 4);
  conv(wA, wE + 4 * 147456, emb_b + 4 * 128, wB, 106, 128, 128, 1, 4);
  conv(wB, wE + 5 * 147456, emb_b + 5 * 128, wA, 104, 128, 128, 0, 4);
  {
    int n = 102 * 102 * 128;   // max over spp -> E (HWC) in wX
    k_max_spp<<<(n + 255) / 256, 256, 0, stream>>>(wA, wX, n, n);
  }
  // pixel-feature chain (batch = 1)
  conv(wX, wP0, pf_b0, wB, 102, 128, 128, 1, 1);
  conv(wB, wP1, pf_b1, wX, 100, 128, 128, 1, 1);
  conv(wX, wP2, pf_b2, wPf, 98, 128, 64, 0, 1);

  // kernel-prediction MLP over 921600 sites: 57600 waves, 8 waves/block
  k_kp_mlp<<<7200, 256, 0, stream>>>(features, radiance, wPf, sf_w, sf_b,
                                     wK0, kp_b0, wK1, kp_b1, kp_w2, kp_b2, wUn, wSw);
  k_finalize<<<(27648 + 255) / 256, 256, 0, stream>>>(wUn, wSw, out);
}
